// LSTM_60516089200778
// MI455X (gfx1250) — matrix operand order repaired and checked by reading, not compiled, not run on hardware
//
#include <hip/hip_runtime.h>

typedef __attribute__((ext_vector_type(16))) __bf16 v16bf;
typedef __attribute__((ext_vector_type(8)))  float  v8f;

namespace {

constexpr int kB  = 256;   // batch
constexpr int kT  = 2048;  // timesteps
constexpr int kH  = 128;   // hidden
constexpr int kMT = 16;    // batch rows per workgroup (WMMA M)
constexpr int kXC = 256;   // x staging chunk (timesteps)

union AFrag { v16bf v; uint4 u[2]; };

__device__ __forceinline__ float fast_sigmoid(float g) {
  // 1/(1+exp(-g)) on native v_exp_f32 / v_rcp_f32
  float e = __builtin_amdgcn_exp2f(-1.4426950408889634f * g);
  return __builtin_amdgcn_rcpf(1.0f + e);
}

// Split workgroup barrier halves (verbatim opcodes from gfx1250 codegen).
__device__ __forceinline__ void wg_barrier_signal_after_lds() {
  asm volatile("s_wait_dscnt 0x0\n\ts_barrier_signal -1" ::: "memory");
}
__device__ __forceinline__ void wg_barrier_wait() {
  asm volatile("s_barrier_wait -1" ::: "memory");
}

} // namespace

__global__ __launch_bounds__(256, 1)
void lstm_wmma_kernel(const float* __restrict__ x,
                      const float* __restrict__ W,
                      const float* __restrict__ b,
                      float* __restrict__ out)
{
  // h_{t} shared across the 8 waves, bf16 hi/lo split, double buffered: 16 KB
  __shared__ __align__(16) __bf16 h_hi[2][kMT][kH];
  __shared__ __align__(16) __bf16 h_lo[2][kMT][kH];
  // x staged transposed [t mod 256][row]: 16 KB
  __shared__ __align__(16) float  x_stage[kXC][kMT];

  const int tid  = threadIdx.x;
  const int wave = tid >> 5;          // 0..7 -> N-tile
  const int lane = tid & 31;
  const int wg   = blockIdx.x;        // 0..15 -> batch tile

  const int n     = wave * 16 + (lane & 15);  // output column (0..127)
  const int half  = lane >> 4;                // lane half (0/1)
  const int m     = lane & 15;                // A-frag row
  const int mbase = half * 8;                 // C/D frag row base

  // ---- zero h0 in the t=0 read buffer (buffer 1) ----
  for (int i = tid; i < kMT * kH; i += 256) {
    (&h_hi[1][0][0])[i] = (__bf16)0.0f;
    (&h_lo[1][0][0])[i] = (__bf16)0.0f;
  }

  // ---- build register-resident B-matrix fragments of W (bf16 hi/lo) ----
  // B layout as built here (contiguous k order, same as the A loads): lanes 0-15 hold col n, K=0..15; lanes 16-31 K=16..31
  v16bf w_hi[4], w_lo[4];
  #pragma unroll
  for (int kc = 0; kc < 4; ++kc) {
    #pragma unroll
    for (int j = 0; j < 16; ++j) {
      int k = kc * 32 + half * 16 + j;
      float wv = W[k * kH + n];
      __bf16 hi = (__bf16)wv;
      w_hi[kc][j] = hi;
      w_lo[kc][j] = (__bf16)(wv - (float)hi);
    }
  }
  const float wlast = W[kH * kH + n];  // W[128][n]: weight of x_t
  const float bcol  = b[n];

  v8f cst = {};  // fp32 cell state for rows mbase..mbase+7, column n
  float* outp = out + (size_t)(wg * kMT + mbase) * kT * kH + n;
  const size_t rstr = (size_t)kT * kH;  // per-row stride in out

  __syncthreads();              // init visible (complete barrier round)
  asm volatile("s_barrier_signal -1" ::: "memory");  // prime split protocol

  for (int t = 0; t < kT; ++t) {
    const int wb = t & 1, rb = wb ^ 1;
    wg_barrier_wait();  // all waves finished previous step (reads + h stores)

    // ---- refill x staging chunk every 256 steps (uniform control flow) ----
    if ((t & (kXC - 1)) == 0) {
      for (int i = tid; i < kMT * kXC; i += 256) {
        int row = i >> 8;          // i / kXC
        int tt  = i & (kXC - 1);
        x_stage[tt][row] = x[(size_t)(wg * kMT + row) * kT + t + tt];
      }
      if (t + kXC < kT) {          // prefetch next chunk into cache
        int prow = tid >> 4;               // 0..15
        int ptt  = (tid & 15) * 16;        // 64B-spaced points per row
        __builtin_prefetch(&x[(size_t)(wg * kMT + prow) * kT + t + kXC + ptt], 0, 1);
      }
      __syncthreads();           // self-contained barrier round
    }

    // ---- A fragments of h_{t-1} from LDS (two contiguous b128 loads each) ----
    AFrag ahi[4], alo[4];
    const __bf16* rhi = &h_hi[rb][m][0];
    const __bf16* rlo = &h_lo[rb][m][0];
    #pragma unroll
    for (int kc = 0; kc < 4; ++kc) {
      int k0 = kc * 32 + half * 16;
      ahi[kc].u[0] = *(const uint4*)(rhi + k0);
      ahi[kc].u[1] = *(const uint4*)(rhi + k0 + 8);
      alo[kc].u[0] = *(const uint4*)(rlo + k0);
      alo[kc].u[1] = *(const uint4*)(rlo + k0 + 8);
    }

    // ---- g = h @ W via 3 independent chains of 4 bf16 WMMAs (hi/lo split),
    //      all chains start from the inline-0 C operand ----
    v8f acc0 = {}, acc1 = {}, acc2 = {};
    #pragma unroll
    for (int kc = 0; kc < 4; ++kc) {
      acc0 = __builtin_amdgcn_wmma_f32_16x16x32_bf16(
          false, ahi[kc].v, false, w_hi[kc], (short)0, acc0, false, false);
      acc1 = __builtin_amdgcn_wmma_f32_16x16x32_bf16(
          false, ahi[kc].v, false, w_lo[kc], (short)0, acc1, false, false);
      acc2 = __builtin_amdgcn_wmma_f32_16x16x32_bf16(
          false, alo[kc].v, false, w_hi[kc], (short)0, acc2, false, false);
    }

    // ---- rank-1 x-term folded into the reduction (off the chain head) ----
    const float* xr = &x_stage[t & (kXC - 1)][mbase];
    float4 x0 = *(const float4*)xr;
    float4 x1 = *(const float4*)(xr + 4);
    float xv[8] = {x0.x, x0.y, x0.z, x0.w, x1.x, x1.y, x1.z, x1.w};

    // ---- gating, state update, h broadcast ----
    float hn_v[8];
    #pragma unroll
    for (int r = 0; r < 8; ++r) {
      float g  = ((acc0[r] + acc1[r]) + acc2[r]) + __builtin_fmaf(xv[r], wlast, bcol);
      float sg = fast_sigmoid(g);
      float cn = sg * (cst[r] + g);   // sg*c + sg*g
      cst[r]   = cn;
      float hn = sg * cn;
      hn_v[r]  = hn;
      __bf16 hb = (__bf16)hn;
      h_hi[wb][mbase + r][n] = hb;
      h_lo[wb][mbase + r][n] = (__bf16)(hn - (float)hb);
    }

    // h tile is in LDS: signal as soon as DS drains, *then* do global stores
    wg_barrier_signal_after_lds();

    #pragma unroll
    for (int r = 0; r < 8; ++r)
      outp[(size_t)r * rstr] = hn_v[r];   // out[row][t][n], f32
    outp += kH;
  }
}

extern "C" void kernel_launch(void* const* d_in, const int* in_sizes, int n_in,
                              void* d_out, int out_size, void* d_ws, size_t ws_size,
                              hipStream_t stream) {
  const float* x = (const float*)d_in[0];  // [256, 2048]
  const float* W = (const float*)d_in[1];  // [129, 128]
  const float* b = (const float*)d_in[2];  // [128]
  float* out = (float*)d_out;              // [256, 2048, 128]
  (void)in_sizes; (void)n_in; (void)out_size; (void)d_ws; (void)ws_size;

  dim3 grid(kB / kMT);   // 16 persistent workgroups
  dim3 block(256);       // 8 waves: one 16-column N-tile each
  hipLaunchKernelGGL(lstm_wmma_kernel, grid, block, 0, stream, x, W, b, out);
}
